// SAPILayer_70214125355057
// MI455X (gfx1250) — matrix operand order repaired and checked by reading, not compiled, not run on hardware
//
#include <hip/hip_runtime.h>

// Problem constants (match reference)
#define HH    128
#define WW    128
#define DD    32
#define BB    8
#define NN    512
#define EPSC  1e-6f

typedef __attribute__((ext_vector_type(16))) _Float16 v16h;
typedef __attribute__((ext_vector_type(8)))  float    v8f;

__device__ __forceinline__ float phi_fn(float t) {
    // elu(t)+1 : t>0 ? t+1 : exp(t)
    return t > 0.0f ? (t + 1.0f) : __expf(t);
}

// ---------------------------------------------------------------------------
// Stage 1: per-batch linear attention -> per-point weight w[n], plus x[n],y[n]
// One block per batch, 1024 threads (32 waves).
//   Phase A: thread (d1,d2) computes kv[d1][d2] = sum_n kf[n,d1]*v[n,d2]
//   Phase B: threads 0..511 each finish one point n.
// ---------------------------------------------------------------------------
__global__ void __launch_bounds__(1024)
sapi_stage1(const float* __restrict__ diags,
            const float* __restrict__ Wq,
            const float* __restrict__ Wk,
            const float* __restrict__ Wv,
            const float* __restrict__ log_sigma,
            const float* __restrict__ log_lambda,
            float* __restrict__ xbuf,
            float* __restrict__ ybuf,
            float* __restrict__ wbuf)
{
    __shared__ float sx[NN];
    __shared__ float sy[NN];
    __shared__ float kv[DD][DD];
    __shared__ float ksum[DD];

    const int b   = blockIdx.x;
    const int tid = threadIdx.x;

    if (tid < NN) {
        const float* pd = diags + ((size_t)b * NN + tid) * 3;
        float x = pd[1];
        float y = pd[2] - pd[1];
        sx[tid] = x;
        sy[tid] = y;
    }
    __syncthreads();

    // Phase A
    const int d1 = tid >> 5;   // 0..31
    const int d2 = tid & 31;   // 0..31
    const float wk0 = Wk[d1 * 2 + 0], wk1 = Wk[d1 * 2 + 1];
    const float wv0 = Wv[d2 * 2 + 0], wv1 = Wv[d2 * 2 + 1];
    float acc = 0.0f, ks = 0.0f;
    for (int n = 0; n < NN; ++n) {
        float x  = sx[n], y = sy[n];
        float kf = phi_fn(x * wk0 + y * wk1);
        float v  = x * wv0 + y * wv1;
        acc += kf * v;
        ks  += kf;
    }
    kv[d1][d2] = acc;
    if (d2 == 0) ksum[d1] = ks;
    __syncthreads();

    // Phase B
    if (tid < NN) {
        const float x = sx[tid], y = sy[tid];
        float num[DD];
#pragma unroll
        for (int d = 0; d < DD; ++d) num[d] = 0.0f;
        float denom = 0.0f;
        for (int j = 0; j < DD; ++j) {
            float qf = phi_fn(x * Wq[j * 2 + 0] + y * Wq[j * 2 + 1]);
            denom += qf * ksum[j];
#pragma unroll
            for (int d = 0; d < DD; ++d) num[d] += qf * kv[j][d];
        }
        denom += EPSC;
        float ss = 0.0f;
#pragma unroll
        for (int d = 0; d < DD; ++d) ss += num[d] * num[d];
        float lam = fmaxf(__expf(log_lambda[0]), 1e-6f);
        float psi = y * __expf(-y / lam);
        float wn  = (sqrtf(ss) / denom) * psi;
        size_t o  = (size_t)b * NN + tid;
        xbuf[o] = x;
        ybuf[o] = y;
        wbuf[o] = wn;
    }
}

// ---------------------------------------------------------------------------
// Stage 2: separable Gaussian splat as batched GEMM via WMMA.
//   out[b](128x128) = (w .* Gy)^T (128x512) @ Gx (512x128)
// grid = (8 tile-groups, 8 batches), block = 256 (8 waves, 1 tile per wave).
// ---------------------------------------------------------------------------
__global__ void __launch_bounds__(256)
sapi_stage2(const float* __restrict__ xbuf,
            const float* __restrict__ ybuf,
            const float* __restrict__ wbuf,
            const float* __restrict__ log_sigma,
            float* __restrict__ out)
{
    __shared__ float sx[NN];
    __shared__ float sy[NN];
    __shared__ float sw[NN];

    const int b   = blockIdx.y;
    const int tid = threadIdx.x;

    for (int i = tid; i < NN; i += 256) {
        size_t o = (size_t)b * NN + i;
        sx[i] = xbuf[o];
        sy[i] = ybuf[o];
        sw[i] = wbuf[o];
    }
    __syncthreads();

    const float inv2s2 = 0.5f * __expf(-2.0f * log_sigma[0]);  // 1/(2*sigma^2)

    const int wave = tid >> 5;
    const int lane = tid & 31;
    const int tile = blockIdx.x * 8 + wave;   // 0..63
    const int tm   = tile >> 3;               // row tile 0..7
    const int tn   = tile & 7;                // col tile 0..7
    const int lm   = lane & 15;               // M (for A) / N (for B) index
    const int hi   = lane >> 4;               // lane half

    const float step = 1.0f / 127.0f;
    const float yrow = (float)(tm * 16 + lm) * step;  // ys[row] for A rows
    const float xcol = (float)(tn * 16 + lm) * step;  // xs[col] for B cols

    v8f accv = {};

    for (int k0 = 0; k0 < NN; k0 += 32) {
        v16h afrag, bfrag;
#pragma unroll
        for (int e = 0; e < 16; ++e) {
            // A (16x32 f16): elem e, lane-half hi -> K = (e&7) + (e<8?0:16) + 8*hi
            int ka = k0 + (e & 7) + ((e & 8) << 1) + (hi << 3);
            float dy = yrow - sy[ka];
            float av = sw[ka] * __expf(-dy * dy * inv2s2);
            afrag[e] = (_Float16)av;
            // B (32x16 f16): elem e, lane-half hi -> K = (e&7) + (e<8?0:16) + 8*hi, N = lane%16
            int kb = k0 + (e & 7) + ((e & 8) << 1) + (hi << 3);
            float dx = xcol - sx[kb];
            bfrag[e] = (_Float16)__expf(-dx * dx * inv2s2);
        }
        accv = __builtin_amdgcn_wmma_f32_16x16x32_f16(
            false, afrag, false, bfrag, (short)0, accv, false, false);
    }

    // C/D layout: VGPR r -> row = 16*tm + r + 8*hi ; col = 16*tn + lane%16
#pragma unroll
    for (int r = 0; r < 8; ++r) {
        int row = tm * 16 + r + (hi << 3);
        int col = tn * 16 + lm;
        out[(size_t)b * HH * WW + (size_t)row * WW + col] = accv[r];
    }
}

// ---------------------------------------------------------------------------
extern "C" void kernel_launch(void* const* d_in, const int* in_sizes, int n_in,
                              void* d_out, int out_size, void* d_ws, size_t ws_size,
                              hipStream_t stream) {
    const float* diags      = (const float*)d_in[0];
    const float* Wq         = (const float*)d_in[1];
    const float* Wk         = (const float*)d_in[2];
    const float* Wv         = (const float*)d_in[3];
    const float* log_sigma  = (const float*)d_in[4];
    const float* log_lambda = (const float*)d_in[5];
    float* out = (float*)d_out;

    float* ws   = (float*)d_ws;
    float* xbuf = ws;                 // B*N floats
    float* ybuf = ws + BB * NN;       // B*N floats
    float* wbuf = ws + 2 * BB * NN;   // B*N floats

    sapi_stage1<<<dim3(BB), dim3(1024), 0, stream>>>(
        diags, Wq, Wk, Wv, log_sigma, log_lambda, xbuf, ybuf, wbuf);

    sapi_stage2<<<dim3(8, BB), dim3(256), 0, stream>>>(
        xbuf, ybuf, wbuf, log_sigma, out);
}
